// ProbSparseAttention_86285892977192
// MI455X (gfx1250) — compile-verified, matrix operand order checked by reading, not run on hardware
//
#include <hip/hip_runtime.h>
#include <cstddef>

// ---------------------------------------------------------------------------
// ProbSparse attention for MI455X (gfx1250), wave32 + WMMA bf16 + async LDS.
// Pipeline:
//   1) pack Wk/Wv -> bf16
//   2) Qs = gather(x, idx) @ Wq^T * 1/sqrt(DH)   (bf16, padded to 64 rows)
//   3) K = x@Wk^T  -> (B,H,T,DH) bf16 ; V = x@Wv^T -> (B,H,DH,T) bf16 (WMMA GEMM)
//   4) fused flash attention per (b,h): ctx (B,H,64,DH) f32 + mean (B,H,DH)
//      (K/V tiles staged with global_load_async_to_lds_b128)
//   5) project the B*(U+1) distinct rows by Wo (+bo)
//   6) broadcast/scatter rows into (B,T,D) output
// ---------------------------------------------------------------------------

#define DEV static __device__ __forceinline__

typedef __attribute__((ext_vector_type(16))) __bf16 bf16x16;
typedef __attribute__((ext_vector_type(8)))  __bf16 bf16x8;
typedef __attribute__((ext_vector_type(8)))  float  f32x8;
typedef __attribute__((ext_vector_type(4)))  float  f32x4;
typedef __attribute__((ext_vector_type(8)))  unsigned short us8;

constexpr int Bc = 8, Tc = 4096, Dc = 1024, Hc = 16, DHc = 64;
constexpr int UPAD = 64;   // padded query-row count per (b,h)

// native f32 -> bf16 (lowers to v_cvt_pk_bf16_f32 class ops on gfx1250)
DEV unsigned short f2bf(float f) {
  __bf16 h = (__bf16)f;
  return __builtin_bit_cast(unsigned short, h);
}

DEV bf16x16 frag16(const unsigned short* p0, const unsigned short* p1) {
  bf16x8 lo = *(const bf16x8*)p0;   // 16B contiguous
  bf16x8 hi = *(const bf16x8*)p1;   // 16B contiguous
  return __builtin_shufflevector(lo, hi, 0,1,2,3,4,5,6,7,8,9,10,11,12,13,14,15);
}

DEV f32x8 wmma_bf16(bf16x16 a, bf16x16 b, f32x8 c) {
  return __builtin_amdgcn_wmma_f32_16x16x32_bf16(false, a, false, b,
                                                 (short)0, c, false, false);
}

DEV f32x8 zero8() { f32x8 z; for (int i = 0; i < 8; ++i) z[i] = 0.f; return z; }

// CDNA5 async global -> LDS copy (ASYNCcnt tracked). ldsoff is the in-LDS
// byte offset (low 32 bits of the flat shared address).
DEV void async_b128(unsigned ldsoff, const void* g) {
  asm volatile("global_load_async_to_lds_b128 %0, %1, off"
               :: "v"(ldsoff), "v"(g) : "memory");
}
DEV void wait_async0() { asm volatile("s_wait_asynccnt 0" ::: "memory"); }

DEV unsigned lds_off(const void* p) { return (unsigned)(size_t)p; }

// ---------------------------------------------------------------------------
// 1) Weight packing f32 -> bf16
// ---------------------------------------------------------------------------
__global__ void k_pack_w(const float* __restrict__ Wk, const float* __restrict__ Wv,
                         unsigned short* __restrict__ Wkb,
                         unsigned short* __restrict__ Wvb) {
  int i = blockIdx.x * 256 + threadIdx.x;
  if (i < Dc * Dc) { Wkb[i] = f2bf(Wk[i]); Wvb[i] = f2bf(Wv[i]); }
}

__global__ void k_pos_init(int* __restrict__ pos, int U) {
  int i = blockIdx.x * 256 + threadIdx.x;
  if (i < Tc) pos[i] = U;              // default: mean row
}

__global__ void k_pos_scatter(int* __restrict__ pos, const int* __restrict__ idx,
                              int U) {
  int i = threadIdx.x;
  if (i < U) pos[idx[i]] = i;
}

// ---------------------------------------------------------------------------
// 2) Qs projection: B*64 rows, rows j>=U are zero.
//    Qs layout: ((b*H + h)*UPAD + j)*DH + d   (bf16, pre-scaled by 0.125)
// ---------------------------------------------------------------------------
__global__ void __launch_bounds__(256) k_qs(const float* __restrict__ x,
                                            const float* __restrict__ Wq,
                                            const int* __restrict__ idx,
                                            unsigned short* __restrict__ Qs, int U) {
  __shared__ __align__(16) float vec[Dc];
  int r = blockIdx.x;                 // 0 .. B*UPAD-1
  int b = r >> 6, j = r & 63;
  bool valid = (j < U);
  int t = valid ? idx[j] : 0;
  const float* src = x + ((size_t)b * Tc + t) * Dc;
  for (int e = threadIdx.x; e < Dc; e += 256) vec[e] = valid ? src[e] : 0.f;
  __syncthreads();

  int n = blockIdx.y * 256 + threadIdx.x;         // 0..1023
  const f32x4* wrow = (const f32x4*)(Wq + (size_t)n * Dc);
  const f32x4* v4 = (const f32x4*)vec;
  float s = 0.f;
#pragma unroll 8
  for (int k = 0; k < Dc / 4; ++k) {
    f32x4 w = wrow[k]; f32x4 xv = v4[k];
    s += w[0] * xv[0] + w[1] * xv[1] + w[2] * xv[2] + w[3] * xv[3];
  }
  int h = n >> 6, d = n & 63;
  Qs[(((size_t)(b * Hc + h)) * UPAD + j) * DHc + d] = f2bf(s * 0.125f);
}

// ---------------------------------------------------------------------------
// 3) WMMA GEMM: C[m,n] = sum_k x[m,k] * W[n,k], M=B*T, N=D, K=D.
//    BM=128, BN=128, BK=32; 8 waves (4 m-groups x 2 n-groups), 32x64 per wave.
//    blockIdx.z: 0 -> store K as (B,H,T,DH); 1 -> store V as (B,H,DH,T).
// ---------------------------------------------------------------------------
__global__ void __launch_bounds__(256) k_gemm_kv(
    const float* __restrict__ x, const unsigned short* __restrict__ Wkb,
    const unsigned short* __restrict__ Wvb, unsigned short* __restrict__ Kb,
    unsigned short* __restrict__ Vt) {
  __shared__ __align__(16) unsigned short sA[128 * 32];  // 8 KB, row-major (m,k)

  const int tid = threadIdx.x;
  const int lane = tid & 31, w = tid >> 5;
  const int wm = w >> 1, wn = w & 1;
  const int l16 = lane & 15;
  const int kb = (lane >> 4) * 8;      // k-offset / row-half offset (8)
  const int mbase = blockIdx.y * 128, nbase = blockIdx.x * 128;
  const int which = blockIdx.z;
  const unsigned short* W = which ? Wvb : Wkb;

  f32x8 acc[2][4];
  for (int mt = 0; mt < 2; ++mt)
    for (int nt = 0; nt < 4; ++nt) acc[mt][nt] = zero8();

  // staging map: thread -> 16 contiguous elements of the 128x32 A tile
  const int sr = tid >> 1, scol = (tid & 1) * 16;
  const float* asrc = x + ((size_t)(mbase + sr)) * Dc + scol;

  for (int k0 = 0; k0 < Dc; k0 += 32) {
    const f32x4* s4 = (const f32x4*)(asrc + k0);
    f32x4 f0 = s4[0], f1 = s4[1], f2 = s4[2], f3 = s4[3];
    us8 p0, p1;
    p0[0] = f2bf(f0[0]); p0[1] = f2bf(f0[1]); p0[2] = f2bf(f0[2]); p0[3] = f2bf(f0[3]);
    p0[4] = f2bf(f1[0]); p0[5] = f2bf(f1[1]); p0[6] = f2bf(f1[2]); p0[7] = f2bf(f1[3]);
    p1[0] = f2bf(f2[0]); p1[1] = f2bf(f2[1]); p1[2] = f2bf(f2[2]); p1[3] = f2bf(f2[3]);
    p1[4] = f2bf(f3[0]); p1[5] = f2bf(f3[1]); p1[6] = f2bf(f3[2]); p1[7] = f2bf(f3[3]);
    *(us8*)&sA[sr * 32 + scol] = p0;
    *(us8*)&sA[sr * 32 + scol + 8] = p1;
    if (k0 + 32 < Dc) __builtin_prefetch(asrc + k0 + 32, 0, 0);  // global_prefetch
    __syncthreads();

    bf16x16 af[2], bf[4];
    for (int mt = 0; mt < 2; ++mt) {
      int row = wm * 32 + mt * 16 + l16;
      af[mt] = frag16(&sA[row * 32 + kb], &sA[row * 32 + kb + 16]);
    }
    for (int nt = 0; nt < 4; ++nt) {
      int n = nbase + wn * 64 + nt * 16 + l16;
      const unsigned short* p = W + (size_t)n * Dc + k0 + kb;  // contiguous in k
      bf[nt] = frag16(p, p + 16);
    }
    for (int mt = 0; mt < 2; ++mt)
      for (int nt = 0; nt < 4; ++nt)
        acc[mt][nt] = wmma_bf16(af[mt], bf[nt], acc[mt][nt]);
    __syncthreads();
  }

  // store C -> bf16 with layout permutation
  for (int mt = 0; mt < 2; ++mt)
    for (int nt = 0; nt < 4; ++nt)
      for (int v = 0; v < 8; ++v) {
        int row = mbase + wm * 32 + mt * 16 + v + kb;   // m = b*T + t
        int col = nbase + wn * 64 + nt * 16 + l16;      // n = h*DH + d
        int b = row >> 12, t = row & (Tc - 1);
        int h = col >> 6, d = col & 63;
        unsigned short val = f2bf(acc[mt][nt][v]);
        if (which == 0)
          Kb[(((size_t)(b * Hc + h)) * Tc + t) * DHc + d] = val;
        else
          Vt[(((size_t)(b * Hc + h)) * DHc + d) * Tc + t] = val;
      }
}

// ---------------------------------------------------------------------------
// 4) Fused flash attention: 1 WG per (b,h), 4 waves x 16 query rows (64 padded).
//    K/V tiles staged via CDNA5 async global->LDS.
// ---------------------------------------------------------------------------
__global__ void __launch_bounds__(128) k_attn(
    const unsigned short* __restrict__ Qs, const unsigned short* __restrict__ Kb,
    const unsigned short* __restrict__ Vt, float* __restrict__ ctx,
    float* __restrict__ meanv, int U) {
  __shared__ __align__(16) unsigned short sK[64 * 64];       // (t',d)  8 KB
  __shared__ __align__(16) unsigned short sV[64 * 64];       // (d,t')  8 KB
  __shared__ __align__(16) unsigned short sP[4][16 * 64];    // per-wave P, 8 KB
  __shared__ float sC[64 * 64];                              // ctx stage, 16 KB

  const int tid = threadIdx.x, lane = tid & 31, w = tid >> 5;
  const int l16 = lane & 15;
  const int kb = (lane >> 4) * 8;
  const int bh = blockIdx.x;

  const unsigned short* Qbase = Qs + (size_t)bh * UPAD * DHc;
  const unsigned short* Kbase = Kb + (size_t)bh * Tc * DHc;
  const unsigned short* Vbase = Vt + (size_t)bh * DHc * Tc;

  // Q A-fragments (2 k-steps of 32), loaded once
  bf16x16 qf[2];
  {
    int m = w * 16 + l16;
    for (int ks = 0; ks < 2; ++ks) {
      const unsigned short* p = Qbase + m * DHc + ks * 32 + kb;
      qf[ks] = frag16(p, p + 16);
    }
  }

  // per-thread staging addresses (constant over the t-loop except t0 advance)
  const unsigned kDst = lds_off(&sK[tid * 32]);
  const int vd = tid >> 1, vhc = (tid & 1) * 32;
  const unsigned vDst = lds_off(&sV[vd * 64 + vhc]);

  f32x8 o[4]; for (int nt = 0; nt < 4; ++nt) o[nt] = zero8();
  float mrow[8], lrow[8];
  for (int v = 0; v < 8; ++v) { mrow[v] = -1.0e30f; lrow[v] = 0.f; }

  for (int t0 = 0; t0 < Tc; t0 += 64) {
    // ---- async-stage K tile (contiguous) and V^T tile ----
    {
      const unsigned short* src = Kbase + (size_t)t0 * 64 + tid * 32;
      for (int c = 0; c < 4; ++c) async_b128(kDst + c * 16, src + c * 8);
      const unsigned short* vsrc = Vbase + (size_t)vd * Tc + t0 + vhc;
      for (int c = 0; c < 4; ++c) async_b128(vDst + c * 16, vsrc + c * 8);
      wait_async0();
    }
    __syncthreads();

    // ---- S = Q * K^T  (16 x 64 per wave) ----
    f32x8 s[4];
    for (int nt = 0; nt < 4; ++nt) {
      f32x8 a = zero8();
      for (int ks = 0; ks < 2; ++ks) {
        const unsigned short* p = &sK[(nt * 16 + l16) * 64 + ks * 32 + kb];
        a = wmma_bf16(qf[ks], frag16(p, p + 16), a);
      }
      s[nt] = a;
    }

    // ---- online softmax (rows live across 16 lanes per VGPR) ----
    float mnew[8], sc[8];
    for (int v = 0; v < 8; ++v) {
      float mx = fmaxf(fmaxf(s[0][v], s[1][v]), fmaxf(s[2][v], s[3][v]));
      mx = fmaxf(mx, __shfl_xor(mx, 1, 32));
      mx = fmaxf(mx, __shfl_xor(mx, 2, 32));
      mx = fmaxf(mx, __shfl_xor(mx, 4, 32));
      mx = fmaxf(mx, __shfl_xor(mx, 8, 32));
      mnew[v] = fmaxf(mrow[v], mx);
      sc[v] = __expf(mrow[v] - mnew[v]);
      mrow[v] = mnew[v];
      lrow[v] *= sc[v];
    }
    for (int nt = 0; nt < 4; ++nt)
      for (int v = 0; v < 8; ++v) s[nt][v] = __expf(s[nt][v] - mnew[v]);
    for (int v = 0; v < 8; ++v) {
      float ps = s[0][v] + s[1][v] + s[2][v] + s[3][v];
      ps += __shfl_xor(ps, 1, 32);
      ps += __shfl_xor(ps, 2, 32);
      ps += __shfl_xor(ps, 4, 32);
      ps += __shfl_xor(ps, 8, 32);
      lrow[v] += ps;
    }
    for (int nt = 0; nt < 4; ++nt)
      for (int v = 0; v < 8; ++v) o[nt][v] *= sc[v];

    // ---- repack P (C layout -> A layout) through per-wave LDS ----
    unsigned short* P = sP[w];
    for (int nt = 0; nt < 4; ++nt)
      for (int v = 0; v < 8; ++v)
        P[(v + kb) * 64 + nt * 16 + l16] = f2bf(s[nt][v]);
    // same-wave LDS RAW: DS ops are in-order per wave

    // ---- O += P * V ----
    for (int ks = 0; ks < 2; ++ks) {
      const unsigned short* pa = &P[l16 * 64 + ks * 32 + kb];
      bf16x16 af = frag16(pa, pa + 16);
      for (int nt = 0; nt < 4; ++nt) {
        const unsigned short* pb = &sV[(nt * 16 + l16) * 64 + ks * 32 + kb];
        o[nt] = wmma_bf16(af, frag16(pb, pb + 16), o[nt]);
      }
    }
    __syncthreads();
  }

  // ---- epilogue: normalize, write ctx, compute mean over rows < U ----
  float* cb = ctx + (size_t)bh * UPAD * DHc;
  for (int nt = 0; nt < 4; ++nt)
    for (int v = 0; v < 8; ++v) {
      int row = w * 16 + v + kb;
      int col = nt * 16 + l16;
      float val = o[nt][v] / lrow[v];
      cb[row * DHc + col] = val;
      sC[row * 64 + col] = val;
    }
  __syncthreads();
  if (tid < DHc) {
    float ssum = 0.f;
    for (int rr = 0; rr < U; ++rr) ssum += sC[rr * 64 + tid];
    meanv[bh * DHc + tid] = ssum / (float)U;
  }
}

// ---------------------------------------------------------------------------
// 5) Project the B*(U+1) distinct rows by Wo + bo
// ---------------------------------------------------------------------------
__global__ void __launch_bounds__(256) k_rows(const float* __restrict__ ctx,
                                              const float* __restrict__ meanv,
                                              const float* __restrict__ Wo,
                                              const float* __restrict__ bo,
                                              float* __restrict__ ys, int U) {
  __shared__ __align__(16) float vec[Dc];
  int r = blockIdx.x;
  int b = r / (U + 1), j = r % (U + 1);
  for (int e = threadIdx.x; e < Dc; e += 256) {
    int h = e >> 6, d = e & 63;
    vec[e] = (j < U) ? ctx[(((size_t)(b * Hc + h)) * UPAD + j) * DHc + d]
                     : meanv[(b * Hc + h) * DHc + d];
  }
  __syncthreads();

  int n = blockIdx.y * 256 + threadIdx.x;
  const f32x4* wrow = (const f32x4*)(Wo + (size_t)n * Dc);
  const f32x4* v4 = (const f32x4*)vec;
  float s = bo[n];
#pragma unroll 8
  for (int k = 0; k < Dc / 4; ++k) {
    f32x4 ww = wrow[k]; f32x4 xv = v4[k];
    s += ww[0] * xv[0] + ww[1] * xv[1] + ww[2] * xv[2] + ww[3] * xv[3];
  }
  ys[(size_t)r * Dc + n] = s;
}

// ---------------------------------------------------------------------------
// 6) Broadcast/scatter rows into the (B,T,D) output (write-bound, 134 MB)
// ---------------------------------------------------------------------------
__global__ void __launch_bounds__(256) k_scatter(const float* __restrict__ ys,
                                                 const int* __restrict__ pos,
                                                 float* __restrict__ out, int U) {
  int row = blockIdx.x;                 // b*T + t
  int b = row >> 12, t = row & (Tc - 1);
  int p = pos[t];                       // 0..U-1 -> ctx row, U -> mean row
  const f32x4* src = (const f32x4*)(ys + ((size_t)(b * (U + 1) + p)) * Dc);
  f32x4* dst = (f32x4*)(out + (size_t)row * Dc);
  __builtin_nontemporal_store(src[threadIdx.x], &dst[threadIdx.x]);
}

// ---------------------------------------------------------------------------
extern "C" void kernel_launch(void* const* d_in, const int* in_sizes, int n_in,
                              void* d_out, int out_size, void* d_ws, size_t ws_size,
                              hipStream_t stream) {
  (void)n_in; (void)out_size; (void)ws_size;
  const float* x  = (const float*)d_in[0];
  const float* Wq = (const float*)d_in[1];
  const float* Wk = (const float*)d_in[2];
  const float* Wv = (const float*)d_in[3];
  const float* Wo = (const float*)d_in[4];
  const float* bo = (const float*)d_in[5];
  const int* idx  = (const int*)d_in[6];
  int U = in_sizes[6];            // u = 41 for these shapes
  if (U > UPAD) U = UPAD;

  char* ws = (char*)d_ws;
  const size_t MB = 1ull << 20;
  unsigned short* Wkb = (unsigned short*)(ws + 0 * MB);    // 2 MB
  unsigned short* Wvb = (unsigned short*)(ws + 2 * MB);    // 2 MB
  unsigned short* Qsb = (unsigned short*)(ws + 4 * MB);    // 1 MB
  float* meanv        = (float*)(ws + 5 * MB);             // 32 KB
  float* ctx          = (float*)(ws + 6 * MB);             // 2 MB
  float* ys           = (float*)(ws + 8 * MB);             // <4 MB
  int* pos            = (int*)(ws + 12 * MB);              // 16 KB
  unsigned short* Kb  = (unsigned short*)(ws + 16 * MB);   // 64 MiB
  unsigned short* Vt  = (unsigned short*)(ws + 80 * MB);   // 64 MiB  (total 144 MiB)

  k_pack_w<<<(Dc * Dc + 255) / 256, 256, 0, stream>>>(Wk, Wv, Wkb, Wvb);
  k_pos_init<<<Tc / 256, 256, 0, stream>>>(pos, U);
  k_pos_scatter<<<1, UPAD, 0, stream>>>(pos, idx, U);
  k_qs<<<dim3(Bc * UPAD, Dc / 256), 256, 0, stream>>>(x, Wq, idx, Qsb, U);
  k_gemm_kv<<<dim3(Dc / 128, (Bc * Tc) / 128, 2), 256, 0, stream>>>(x, Wkb, Wvb, Kb, Vt);
  k_attn<<<Bc * Hc, 128, 0, stream>>>(Qsb, Kb, Vt, ctx, meanv, U);
  k_rows<<<dim3(Bc * (U + 1), Dc / 256), 256, 0, stream>>>(ctx, meanv, Wo, bo, ys, U);
  k_scatter<<<Bc * Tc, 256, 0, stream>>>(ys, pos, (float*)d_out, U);
}
